// _NonLocalBlockND_49349174231218
// MI455X (gfx1250) — matrix operand order repaired and checked by reading, not compiled, not run on hardware
//
#include <hip/hip_runtime.h>
#include <hip/hip_bf16.h>
#include <math.h>

typedef __bf16 bf16;
typedef __attribute__((ext_vector_type(16))) __bf16 v16bf;
typedef __attribute__((ext_vector_type(8)))  float  v8f;
typedef __attribute__((ext_vector_type(4)))  unsigned int u32x4;

// Problem constants
constexpr int Bc  = 4;
constexpr int Cc  = 512;
constexpr int CIc = 256;
constexpr int Tc  = 16, Hc = 28, Wc = 28;
constexpr int Nc  = Tc * Hc * Wc;          // 12544
constexpr int TPc = 8, HPc = 14, WPc = 14;
constexpr int NPc = TPc * HPc * WPc;       // 1568
constexpr float EPSc = 1e-5f;

// ---------------------------------------------------------------------------
// Fragment loader: two contiguous 8 x bf16 groups (16B each) -> v16bf
// ---------------------------------------------------------------------------
__device__ __forceinline__ v16bf load_frag16(const bf16* p0, const bf16* p1) {
  union U { u32x4 u; bf16 h[8]; };
  U a, b;
  a.u = *reinterpret_cast<const u32x4*>(p0);
  b.u = *reinterpret_cast<const u32x4*>(p1);
  v16bf r;
#pragma unroll
  for (int i = 0; i < 8; ++i) { r[i] = a.h[i]; r[i + 8] = b.h[i]; }
  return r;
}

__device__ __forceinline__ v8f wmma_bf16(v16bf a, v16bf b, v8f c) {
  return __builtin_amdgcn_wmma_f32_16x16x32_bf16(false, a, false, b,
                                                 (short)0, c, false, false);
}

// ---------------------------------------------------------------------------
// CDNA5 async global->LDS copy (ASYNCcnt-tracked, bypasses VGPRs).
// LDS address = low 32 bits of the generic pointer (== addrspace(3) offset).
// ---------------------------------------------------------------------------
__device__ __forceinline__ unsigned lds_off_u32(const void* p) {
  return (unsigned)(unsigned long long)p;
}
__device__ __forceinline__ void async_copy_b128(const void* lds_dst,
                                                const void* gsrc) {
  asm volatile("global_load_async_to_lds_b128 %0, %1, off"
               :: "v"(lds_off_u32(lds_dst)), "v"(gsrc)
               : "memory");
}
__device__ __forceinline__ void wait_asynccnt0() {
  asm volatile("s_wait_asynccnt 0" ::: "memory");
}

// ---------------------------------------------------------------------------
// Generic 1x1x1-conv GEMM:  out[b] = Wt(OxK) * Bsrc[b](KxNn) + bias
//   OUT_MODE 0: fp32 out, (o, n) layout       (phi/g pre-pool, W_y)
//   OUT_MODE 1: bf16 out, (n, o) layout       (theta, ready as QK A-operand)
//   BSRC_F32  : B source is fp32 (x) vs bf16 (y)
//   BSRC_KMAJ : B source is (K, Nn) row-major (x) vs (Nn, K) row-major (y)
// Block: 256 threads (8 waves). Tile: 128(M) x 64(N) x 32(K).
// ---------------------------------------------------------------------------
template <int OUT_MODE, int BSRC_F32, int BSRC_KMAJ>
__global__ __launch_bounds__(256) void k_conv_gemm(
    const float* __restrict__ Wt, const float* __restrict__ bias,
    const void* __restrict__ Bv, long b_bstride,
    float* __restrict__ outF, bf16* __restrict__ outH, long o_bstride,
    int O, int K, int Nn) {
  constexpr int BM = 128, BN = 64, BK = 32, PAD = 8;
  __shared__ bf16 As[BM][BK + PAD];
  __shared__ bf16 Bs[BN][BK + PAD];

  const int tid  = threadIdx.x;
  const int wave = tid >> 5;
  const int lane = tid & 31;
  const int batch = blockIdx.z;
  const int m0 = blockIdx.y * BM;
  const int n0 = blockIdx.x * BN;

  v8f acc[4];
#pragma unroll
  for (int t = 0; t < 4; ++t)
#pragma unroll
    for (int v = 0; v < 8; ++v) acc[t][v] = 0.f;

  for (int k0 = 0; k0 < K; k0 += BK) {
    // ---- stage A (weights, fp32 -> bf16), [m][k] layout
    {
      const int m  = tid >> 1;
      const int kh = (tid & 1) * 16;
      const float* src = Wt + (long)(m0 + m) * K + k0 + kh;
#pragma unroll
      for (int i = 0; i < 16; ++i) As[m][kh + i] = (bf16)src[i];
    }
    // ---- stage B, [n][k] layout so B-fragments are k-contiguous
    if (BSRC_KMAJ) {
      const float* src = (const float*)Bv + (long)batch * b_bstride;
#pragma unroll
      for (int i = 0; i < 8; ++i) {
        const int e = tid * 8 + i;
        const int k = e >> 6;      // /64
        const int n = e & 63;
        Bs[n][k] = (bf16)src[(long)(k0 + k) * Nn + n0 + n];
      }
      if (k0 + BK < K)  // gfx1250 global_prefetch_b8 of next B tile
        __builtin_prefetch((const float*)Bv + (long)batch * b_bstride +
                               (long)(k0 + BK) * Nn + n0, 0, 1);
    } else {
      const bf16* src = (const bf16*)Bv + (long)batch * b_bstride;
      const int n  = tid >> 2;
      const int kq = (tid & 3) * 8;
      const bf16* p = src + (long)(n0 + n) * K + k0 + kq;
#pragma unroll
      for (int i = 0; i < 8; ++i) Bs[n][kq + i] = p[i];
    }
    __syncthreads();

    // ---- compute: wave handles rows [wave*16, wave*16+16) x 64 cols
    {
      const int mw  = wave * 16 + (lane & 15);
      const int kbA = (lane < 16) ? 0 : 8;
      const v16bf af = load_frag16(&As[mw][kbA], &As[mw][kbA + 16]);
      const int kbB = (lane < 16) ? 0 : 8;
#pragma unroll
      for (int t = 0; t < 4; ++t) {
        const int nn = t * 16 + (lane & 15);
        const v16bf bf = load_frag16(&Bs[nn][kbB], &Bs[nn][kbB + 16]);
        acc[t] = wmma_bf16(af, bf, acc[t]);
      }
    }
    __syncthreads();
  }

  // ---- epilogue
  const int lhalf = (lane < 16) ? 0 : 8;
  float bval[8];
#pragma unroll
  for (int v = 0; v < 8; ++v) bval[v] = bias[m0 + wave * 16 + lhalf + v];
#pragma unroll
  for (int t = 0; t < 4; ++t) {
    const int nn = n0 + t * 16 + (lane & 15);
#pragma unroll
    for (int v = 0; v < 8; ++v) {
      const int mm = m0 + wave * 16 + lhalf + v;
      const float val = acc[t][v] + bval[v];
      if (OUT_MODE == 0)
        outF[(long)batch * o_bstride + (long)mm * Nn + nn] = val;
      else
        outH[(long)batch * o_bstride + (long)nn * O + mm] = (bf16)val;
    }
  }
}

// ---------------------------------------------------------------------------
// MaxPool3d(2,2,2) on (B, CI, T, H, W) fp32, writing bf16.
//   GMODE 0: (b, ci, n')  -- used for g  (PV B-operand: key-contiguous rows)
//   GMODE 1: (b, n', ci)  -- used for phi (QK B-operand: ci-contiguous rows)
// ---------------------------------------------------------------------------
template <int GMODE>
__global__ void k_pool(const float* __restrict__ in, bf16* __restrict__ out) {
  const long idx = (long)blockIdx.x * 256 + threadIdx.x;
  const long total = (long)Bc * CIc * NPc;
  if (idx >= total) return;
  const int np = (int)(idx % NPc);
  const long r = idx / NPc;
  const int ci = (int)(r % CIc);
  const int b  = (int)(r / CIc);
  const int wp = np % WPc;
  const int hp = (np / WPc) % HPc;
  const int tp = np / (WPc * HPc);
  const float* base = in + (((long)b * CIc + ci) * Tc + tp * 2) * (Hc * Wc) +
                      (hp * 2) * Wc + wp * 2;
  float m = -__builtin_inff();
#pragma unroll
  for (int dt = 0; dt < 2; ++dt)
#pragma unroll
    for (int dh = 0; dh < 2; ++dh)
#pragma unroll
      for (int dw = 0; dw < 2; ++dw)
        m = fmaxf(m, base[dt * Hc * Wc + dh * Wc + dw]);
  if (GMODE == 0)
    out[((long)b * CIc + ci) * NPc + np] = (bf16)m;
  else
    out[((long)b * NPc + np) * CIc + ci] = (bf16)m;
}

// ---------------------------------------------------------------------------
// Flash attention: y = softmax(theta @ phi^T-equivalent) @ g, fused.
// Block = 128 threads (4 waves); 64 queries/WG; 32-key streaming chunks.
// All global->LDS staging via CDNA5 async b128 copies (ASYNCcnt).
//   thetaB: (b, N, CI)   phiNB: (b, NP, CI)   gT: (b, CI, NP)
// ---------------------------------------------------------------------------
__global__ __launch_bounds__(128) void k_attention(
    const bf16* __restrict__ thetaB, const bf16* __restrict__ phiNB,
    const bf16* __restrict__ gT, bf16* __restrict__ yB) {
  constexpr int QT = 64, KT = 32, PAD = 8;
  __shared__ bf16 Qs[QT][CIc + PAD];         // [query][ci]
  __shared__ bf16 Ks[KT][CIc + PAD];         // [key][ci]   (B-op of QK^T)
  __shared__ bf16 Vs[CIc][KT + PAD];         // [ci][key]   (B-op of PV)
  __shared__ bf16 Ps[4][16][KT + PAD];       // per-wave P staging (A-op of PV)

  const int tid  = threadIdx.x;
  const int wave = tid >> 5;
  const int lane = tid & 31;
  const int b  = blockIdx.y;
  const int q0 = blockIdx.x * QT;
  const int n1 = lane & 15;
  const int lh = (lane < 16) ? 0 : 8;

  // ---- async-stage Q tile (64 x 256 bf16): 2048 x 16B chunks
  {
    const bf16* src = thetaB + ((long)b * Nc + q0) * CIc;
    for (int i = tid; i < QT * (CIc / 8); i += 128) {
      const int m  = i >> 5;           // / (256/8)
      const int c8 = (i & 31) * 8;
      async_copy_b128(&Qs[m][c8], src + (long)m * CIc + c8);
    }
  }

  v8f Oa[16];
#pragma unroll
  for (int t = 0; t < 16; ++t)
#pragma unroll
    for (int v = 0; v < 8; ++v) Oa[t][v] = 0.f;
  float mrow[8], lrow[8];
#pragma unroll
  for (int v = 0; v < 8; ++v) { mrow[v] = -__builtin_inff(); lrow[v] = 0.f; }

  const int mq = wave * 16 + n1;

  for (int kb0 = 0; kb0 < NPc; kb0 += KT) {
    __syncthreads();  // all waves done reading previous Ks/Vs
    // ---- async-stage K chunk: phiNB rows (np, ci) -> Ks[key][ci]
    {
      const bf16* srcK = phiNB + ((long)b * NPc + kb0) * CIc;
      for (int i = tid; i < KT * (CIc / 8); i += 128) {
        const int k  = i >> 5;
        const int c8 = (i & 31) * 8;
        async_copy_b128(&Ks[k][c8], srcK + (long)k * CIc + c8);
      }
    }
    // ---- async-stage V chunk: gT rows (ci, np) -> Vs[ci][key]
    {
      const bf16* srcV = gT + (long)b * CIc * NPc + kb0;
      for (int i = tid; i < CIc * (KT / 8); i += 128) {
        const int cidx = i >> 2;
        const int k8   = (i & 3) * 8;
        async_copy_b128(&Vs[cidx][k8], srcV + (long)cidx * NPc + k8);
      }
    }
    wait_asynccnt0();   // this wave's async copies (incl. Q on iter 0) landed
    __syncthreads();    // all waves' copies landed

    // ---- S = Q x K^T (16 queries x 32 keys per wave)
    v8f S0, S1;
#pragma unroll
    for (int v = 0; v < 8; ++v) { S0[v] = 0.f; S1[v] = 0.f; }
#pragma unroll
    for (int kk = 0; kk < CIc; kk += 32) {
      const int kbA = kk + ((lane < 16) ? 0 : 8);
      const v16bf af = load_frag16(&Qs[mq][kbA], &Qs[mq][kbA + 16]);
      const int kbB = kk + ((lane < 16) ? 0 : 8);
      const v16bf b0 = load_frag16(&Ks[n1][kbB], &Ks[n1][kbB + 16]);
      const v16bf b1 = load_frag16(&Ks[16 + n1][kbB], &Ks[16 + n1][kbB + 16]);
      S0 = wmma_bf16(af, b0, S0);
      S1 = wmma_bf16(af, b1, S1);
    }

    // ---- online softmax (row stats across the 16-lane half-groups)
    float scl[8];
#pragma unroll
    for (int v = 0; v < 8; ++v) {
      float cm = fmaxf(S0[v], S1[v]);
#pragma unroll
      for (int off = 8; off >= 1; off >>= 1)
        cm = fmaxf(cm, __shfl_xor(cm, off, 16));
      const float mn = fmaxf(mrow[v], cm);
      const float sc = __expf(mrow[v] - mn);
      const float p0 = __expf(S0[v] - mn);
      const float p1 = __expf(S1[v] - mn);
      S0[v] = p0; S1[v] = p1;
      float rs = p0 + p1;
#pragma unroll
      for (int off = 8; off >= 1; off >>= 1) rs += __shfl_xor(rs, off, 16);
      lrow[v] = lrow[v] * sc + rs;
      mrow[v] = mn;
      scl[v]  = sc;
    }
#pragma unroll
    for (int t = 0; t < 16; ++t)
#pragma unroll
      for (int v = 0; v < 8; ++v) Oa[t][v] *= scl[v];

    // ---- P (C-layout) -> LDS -> A-fragment layout
#pragma unroll
    for (int v = 0; v < 8; ++v) {
      const int r = lh + v;
      Ps[wave][r][n1]      = (bf16)S0[v];
      Ps[wave][r][16 + n1] = (bf16)S1[v];
    }
    __syncthreads();  // cross-lane LDS visibility (uniform across waves)

    // ---- O += P(16x32) x V(32x256): one WMMA per ci-tile this chunk
    const int kbP = (lane < 16) ? 0 : 8;
    const v16bf pf =
        load_frag16(&Ps[wave][n1][kbP], &Ps[wave][n1][kbP + 16]);
    const int kbV = (lane < 16) ? 0 : 8;
#pragma unroll
    for (int t = 0; t < 16; ++t) {
      const int nc = t * 16 + n1;
      const v16bf vf = load_frag16(&Vs[nc][kbV], &Vs[nc][kbV + 16]);
      Oa[t] = wmma_bf16(pf, vf, Oa[t]);
    }
  }

  // ---- epilogue: O /= rowsum, write y (b, n, ci) bf16
  float invl[8];
#pragma unroll
  for (int v = 0; v < 8; ++v) invl[v] = 1.f / lrow[v];
#pragma unroll
  for (int t = 0; t < 16; ++t) {
#pragma unroll
    for (int v = 0; v < 8; ++v) {
      const int r = q0 + wave * 16 + lh + v;
      yB[((long)b * Nc + r) * CIc + t * 16 + n1] = (bf16)(Oa[t][v] * invl[v]);
    }
  }
}

// ---------------------------------------------------------------------------
// BatchNorm stats: per channel mean and rsqrt(var+eps) over (b, t, h, w)
// ---------------------------------------------------------------------------
__global__ __launch_bounds__(256) void k_stats(const float* __restrict__ Wy,
                                               float* __restrict__ stats) {
  const int c = blockIdx.x;
  float s = 0.f, sq = 0.f;
  for (int i = threadIdx.x; i < Bc * Nc; i += 256) {
    const int b = i / Nc, n = i % Nc;
    const float v = Wy[((long)b * Cc + c) * Nc + n];
    s += v; sq += v * v;
  }
  __shared__ float ls[256], lq[256];
  ls[threadIdx.x] = s; lq[threadIdx.x] = sq;
  __syncthreads();
  for (int o = 128; o; o >>= 1) {
    if (threadIdx.x < o) {
      ls[threadIdx.x] += ls[threadIdx.x + o];
      lq[threadIdx.x] += lq[threadIdx.x + o];
    }
    __syncthreads();
  }
  if (threadIdx.x == 0) {
    const float inv_n = 1.f / (float)(Bc * Nc);
    const float mean = ls[0] * inv_n;
    const float var  = lq[0] * inv_n - mean * mean;
    stats[c]      = mean;
    stats[Cc + c] = rsqrtf(var + EPSc);
  }
}

// ---------------------------------------------------------------------------
// out = (Wy - mean) * inv_std * gamma + beta + x
// ---------------------------------------------------------------------------
__global__ void k_bn_res(const float* __restrict__ Wy,
                         const float* __restrict__ x,
                         const float* __restrict__ stats,
                         const float* __restrict__ gamma,
                         const float* __restrict__ beta,
                         float* __restrict__ out) {
  const long idx = (long)blockIdx.x * 256 + threadIdx.x;
  const long total = (long)Bc * Cc * Nc;
  if (idx >= total) return;
  const int c = (int)((idx / Nc) % Cc);
  out[idx] = (Wy[idx] - stats[c]) * stats[Cc + c] * gamma[c] + beta[c] + x[idx];
}

// ---------------------------------------------------------------------------
extern "C" void kernel_launch(void* const* d_in, const int* in_sizes, int n_in,
                              void* d_out, int out_size, void* d_ws,
                              size_t ws_size, hipStream_t stream) {
  const float* x       = (const float*)d_in[0];
  const float* g_w     = (const float*)d_in[1];
  const float* g_b     = (const float*)d_in[2];
  const float* theta_w = (const float*)d_in[3];
  const float* theta_b = (const float*)d_in[4];
  const float* phi_w   = (const float*)d_in[5];
  const float* phi_b   = (const float*)d_in[6];
  const float* W_w     = (const float*)d_in[7];
  const float* W_b     = (const float*)d_in[8];
  const float* bn_g    = (const float*)d_in[9];
  const float* bn_b    = (const float*)d_in[10];
  float* out = (float*)d_out;

  // workspace carve-up (256B aligned)
  char* ws = (char*)d_ws;
  size_t off = 0;
  auto carve = [&](size_t bytes) {
    char* p = ws + off;
    off = (off + bytes + 255) & ~(size_t)255;
    return p;
  };
  bf16*  thetaB  = (bf16*)carve((size_t)Bc * Nc * CIc * sizeof(bf16));
  float* convtmp = (float*)carve((size_t)Bc * CIc * Nc * sizeof(float));
  bf16*  phiNB   = (bf16*)carve((size_t)Bc * NPc * CIc * sizeof(bf16)); // (n',ci)
  bf16*  gT      = (bf16*)carve((size_t)Bc * CIc * NPc * sizeof(bf16)); // (ci,n')
  bf16*  yB      = (bf16*)carve((size_t)Bc * Nc * CIc * sizeof(bf16));
  float* Wy      = (float*)carve((size_t)Bc * Cc * Nc * sizeof(float));
  float* stats   = (float*)carve((size_t)2 * Cc * sizeof(float));

  const long xb_stride  = (long)Cc * Nc;   // fp32 elems per batch of x
  const long ci_stride  = (long)CIc * Nc;  // fp32 elems per batch (o,n) out
  const long nci_stride = (long)Nc * CIc;  // bf16 elems per batch (n,ci)
  const long wy_stride  = (long)Cc * Nc;

  dim3 blk256(256), blk128(128);
  dim3 g_ci(Nc / 64, CIc / 128, Bc);   // (196, 2, 4)
  dim3 g_c (Nc / 64, Cc  / 128, Bc);   // (196, 4, 4)

  // theta: bf16, transposed (n, ci)
  k_conv_gemm<1, 1, 1><<<g_ci, blk256, 0, stream>>>(
      theta_w, theta_b, x, xb_stride, nullptr, thetaB, nci_stride,
      CIc, Cc, Nc);
  // phi conv -> fp32 tmp -> pool -> (n', ci) bf16 (async-copy friendly K rows)
  k_conv_gemm<0, 1, 1><<<g_ci, blk256, 0, stream>>>(
      phi_w, phi_b, x, xb_stride, convtmp, nullptr, ci_stride, CIc, Cc, Nc);
  {
    const long tot = (long)Bc * CIc * NPc;
    k_pool<1><<<dim3((tot + 255) / 256), blk256, 0, stream>>>(convtmp, phiNB);
  }
  // g conv -> fp32 tmp -> pool -> (ci, n') bf16 (async-copy friendly V rows)
  k_conv_gemm<0, 1, 1><<<g_ci, blk256, 0, stream>>>(
      g_w, g_b, x, xb_stride, convtmp, nullptr, ci_stride, CIc, Cc, Nc);
  {
    const long tot = (long)Bc * CIc * NPc;
    k_pool<0><<<dim3((tot + 255) / 256), blk256, 0, stream>>>(convtmp, gT);
  }
  // fused softmax(QK^T)V with async LDS staging
  k_attention<<<dim3(Nc / 64, Bc), blk128, 0, stream>>>(thetaB, phiNB, gT, yB);
  // W conv: (C x CI) x y^T -> Wy fp32 (c, n)
  k_conv_gemm<0, 0, 0><<<g_c, blk256, 0, stream>>>(
      W_w, W_b, yB, nci_stride, Wy, nullptr, wy_stride, Cc, CIc, Nc);
  // BN stats + normalize + residual
  k_stats<<<dim3(Cc), blk256, 0, stream>>>(Wy, stats);
  {
    const long tot = (long)Bc * Cc * Nc;
    k_bn_res<<<dim3((tot + 255) / 256), blk256, 0, stream>>>(
        Wy, x, stats, bn_g, bn_b, out);
  }
}
